// BakaAttentionQKV_28552942584360
// MI455X (gfx1250) — compile-verified, matrix operand order checked by reading, not run on hardware
//
#include <hip/hip_runtime.h>
#include <hip/hip_bf16.h>

typedef __attribute__((ext_vector_type(16))) __bf16     v16bf;
typedef __attribute__((ext_vector_type(8)))  float      v8f;
typedef __attribute__((ext_vector_type(4)))  unsigned   u32x4;
typedef __attribute__((ext_vector_type(4)))  float      f32x4;

#define HIDDEN   2048
#define QK_DIM   2048
#define NHEADS   16
#define HEAD256  256
#define Bsz      4
#define Tlen     2048
#define Mdim     (Bsz * Tlen)      // 8192
#define Kdim     2048
#define Ndim     2048
#define EPSV     1e-5f

union FragU { v16bf v; u32x4 q[2]; };

// ---------------------------------------------------------------------------
// fp32 -> bf16 conversion, 8 elements/thread
// ---------------------------------------------------------------------------
__global__ __launch_bounds__(256) void cvt_bf16_kernel(const float* __restrict__ x,
                                                       __bf16* __restrict__ y, int n8) {
    int i = blockIdx.x * 256 + threadIdx.x;
    if (i >= n8) return;
    f32x4 a = ((const f32x4*)x)[2 * i];
    f32x4 b = ((const f32x4*)x)[2 * i + 1];
    union { __bf16 h[8]; u32x4 u; } o;
#pragma unroll
    for (int r = 0; r < 4; ++r) { o.h[r] = (__bf16)a[r]; o.h[4 + r] = (__bf16)b[r]; }
    ((u32x4*)y)[i] = o.u;
}

// ---------------------------------------------------------------------------
// bf16 WMMA GEMM:  C[M,N] = A[M,K] * W[N,K]^T   (both K-major, WMMA-friendly)
// block = 8 waves, wave tile 64x64 (4x4 subtiles of 16x16), block tile 128x256
// mode 0: C row-major fp32 [M,N]      (q / k intermediates)
// mode 1: permuted store for v: out[((b*16+h)*T + t)*128 + d]
// ---------------------------------------------------------------------------
__global__ __launch_bounds__(256) void gemm_bf16_wmma(const __bf16* __restrict__ A,
                                                      const __bf16* __restrict__ W,
                                                      float* __restrict__ C, int mode) {
    const int lane = threadIdx.x & 31;
    const int wave = threadIdx.x >> 5;
    const int wm = wave >> 2;                 // 0..1 along M
    const int wn = wave & 3;                  // 0..3 along N
    const int m0 = blockIdx.y * 128 + wm * 64;
    const int n0 = blockIdx.x * 256 + wn * 64;
    const int mr  = lane & 15;
    const int sel = (lane >> 4) & 1;          // which K-octet this half-wave holds

    v8f acc[4][4] = {};

    const __bf16* Ar[4];
    const __bf16* Br[4];
#pragma unroll
    for (int i = 0; i < 4; ++i) {
        Ar[i] = A + (size_t)(m0 + i * 16 + mr) * Kdim + sel * 8;
        Br[i] = W + (size_t)(n0 + i * 16 + mr) * Kdim + sel * 8;
    }

    for (int k0 = 0; k0 < Kdim; k0 += 32) {
        FragU a[4], b[4];
#pragma unroll
        for (int i = 0; i < 4; ++i) {
            a[i].q[0] = *(const u32x4*)(Ar[i] + k0);
            a[i].q[1] = *(const u32x4*)(Ar[i] + k0 + 16);
            b[i].q[0] = *(const u32x4*)(Br[i] + k0);
            b[i].q[1] = *(const u32x4*)(Br[i] + k0 + 16);
        }
#pragma unroll
        for (int i = 0; i < 4; ++i)
#pragma unroll
            for (int j = 0; j < 4; ++j)
                acc[i][j] = __builtin_amdgcn_wmma_f32_16x16x32_bf16(
                    false, a[i].v, false, b[j].v, (short)0, acc[i][j], false, false);
    }

#pragma unroll
    for (int i = 0; i < 4; ++i) {
        const int rowb = m0 + i * 16 + sel * 8;   // VGPR r holds row rowb+r
#pragma unroll
        for (int j = 0; j < 4; ++j) {
            const int col = n0 + j * 16 + mr;
            if (mode == 0) {
                float* p = C + (size_t)rowb * Ndim + col;
#pragma unroll
                for (int r = 0; r < 8; ++r) p[(size_t)r * Ndim] = acc[i][j][r];
            } else {
                const int h = col >> 7, d = col & 127;
#pragma unroll
                for (int r = 0; r < 8; ++r) {
                    const int rr = rowb + r;
                    const int bb = rr >> 11, tt = rr & 2047;
                    C[(((size_t)bb * NHEADS + h) * Tlen + tt) * 128 + d] = acc[i][j][r];
                }
            }
        }
    }
}

// ---------------------------------------------------------------------------
// Fused concat + LayerNorm + head-split + RoPE for q / k.
// One block (256 thr) per row (b,t); 8 columns/thread.
// LN uses mean == neg/2 exactly; var = mean_f (q_f - neg/2)^2.
// ---------------------------------------------------------------------------
__global__ __launch_bounds__(256) void ln_rope_kernel(const float* __restrict__ Cin,
                                                      const float* __restrict__ gamma,
                                                      const float* __restrict__ beta,
                                                      const float* __restrict__ negp,
                                                      const int*   __restrict__ offp,
                                                      float* __restrict__ out) {
    const int row = blockIdx.x;              // b*T + t
    const int tid = threadIdx.x;
    const int b = row >> 11, t = row & 2047;
    const float neg = negp[0];
    const float c = 0.5f * neg;

    float q[8];
    const f32x4* src = (const f32x4*)(Cin + (size_t)row * 2048);
    f32x4 v0 = src[2 * tid], v1 = src[2 * tid + 1];
#pragma unroll
    for (int r = 0; r < 4; ++r) { q[r] = v0[r]; q[4 + r] = v1[r]; }

    float s = 0.f;
#pragma unroll
    for (int r = 0; r < 8; ++r) { float d = q[r] - c; s += d * d; }
#pragma unroll
    for (int o = 16; o; o >>= 1) s += __shfl_xor(s, o, 32);
    __shared__ float red[8];
    if ((tid & 31) == 0) red[tid >> 5] = s;
    __syncthreads();
    float tot = 0.f;
#pragma unroll
    for (int i = 0; i < 8; ++i) tot += red[i];
    const float rstd = rsqrtf(tot * (1.0f / 2048.0f) + EPSV);

    const int f0 = tid * 8;
    const int d0 = f0 & 255;                 // dim within head (same for both halves)
    const int h1 = f0 >> 8;                  // heads 0..7 (first half), +8 second half
    const float tpos = (float)(t + offp[0]);

    float ox[8], oy[8];
#pragma unroll
    for (int j = 0; j < 4; ++j) {
        const int f = f0 + 2 * j;
        const float u0 = (q[2 * j] - c) * rstd;
        const float u1 = (q[2 * j + 1] - c) * rstd;
        float x0 =  u0 * gamma[f]        + beta[f];
        float x1 =  u1 * gamma[f + 1]    + beta[f + 1];
        float y0 = -u0 * gamma[2048 + f] + beta[2048 + f];
        float y1 = -u1 * gamma[2048 + f + 1] + beta[2048 + f + 1];
        const int dd = d0 + 2 * j;
        if (dd < 128) {                      // rot_dim = 128, interleaved pairs
            const int jj = dd >> 1;
            const float freq = tpos * __expf((float)jj * -0.14391156836f); // 1e4^(-jj/64)
            const float cs = __cosf(freq), sn = __sinf(freq);
            float tx = x0 * cs - x1 * sn; x1 = x1 * cs + x0 * sn; x0 = tx;
            float ty = y0 * cs - y1 * sn; y1 = y1 * cs + y0 * sn; y0 = ty;
        }
        ox[2 * j] = x0; ox[2 * j + 1] = x1;
        oy[2 * j] = y0; oy[2 * j + 1] = y1;
    }

    float* p1 = out + (((size_t)b * NHEADS + h1)     * Tlen + t) * HEAD256 + d0;
    float* p2 = out + (((size_t)b * NHEADS + h1 + 8) * Tlen + t) * HEAD256 + d0;
    ((f32x4*)p1)[0] = f32x4{ox[0], ox[1], ox[2], ox[3]};
    ((f32x4*)p1)[1] = f32x4{ox[4], ox[5], ox[6], ox[7]};
    ((f32x4*)p2)[0] = f32x4{oy[0], oy[1], oy[2], oy[3]};
    ((f32x4*)p2)[1] = f32x4{oy[4], oy[5], oy[6], oy[7]};
}

// ---------------------------------------------------------------------------
extern "C" void kernel_launch(void* const* d_in, const int* in_sizes, int n_in,
                              void* d_out, int out_size, void* d_ws, size_t ws_size,
                              hipStream_t stream) {
    const float* qs = (const float*)d_in[0];
    const float* ks = (const float*)d_in[1];
    const float* vs = (const float*)d_in[2];
    const float* Wq = (const float*)d_in[3];
    const float* Wk = (const float*)d_in[4];
    const float* Wv = (const float*)d_in[5];
    const float* negator = (const float*)d_in[6];
    const float* qg = (const float*)d_in[7];
    const float* qb = (const float*)d_in[8];
    const float* kg = (const float*)d_in[9];
    const float* kb = (const float*)d_in[10];
    const int*   off = (const int*)d_in[11];
    float* out = (float*)d_out;

    const size_t STATE_N = (size_t)Mdim * HIDDEN;     // 16,777,216
    const size_t WEIGHT_N = (size_t)Ndim * Kdim;      //  4,194,304

    char* ws = (char*)d_ws;
    __bf16* qs_h = (__bf16*)ws;               ws += STATE_N * 2;
    __bf16* ks_h = (__bf16*)ws;               ws += STATE_N * 2;
    __bf16* vs_h = (__bf16*)ws;               ws += STATE_N * 2;
    __bf16* Wq_h = (__bf16*)ws;               ws += WEIGHT_N * 2;
    __bf16* Wk_h = (__bf16*)ws;               ws += WEIGHT_N * 2;
    __bf16* Wv_h = (__bf16*)ws;               ws += WEIGHT_N * 2;
    float*  qC   = (float*)ws;                ws += STATE_N * 4;
    float*  kC   = (float*)ws;                ws += STATE_N * 4;

    // 1) convert to bf16
    cvt_bf16_kernel<<<(int)(STATE_N / 8 / 256), 256, 0, stream>>>(qs, qs_h, (int)(STATE_N / 8));
    cvt_bf16_kernel<<<(int)(STATE_N / 8 / 256), 256, 0, stream>>>(ks, ks_h, (int)(STATE_N / 8));
    cvt_bf16_kernel<<<(int)(STATE_N / 8 / 256), 256, 0, stream>>>(vs, vs_h, (int)(STATE_N / 8));
    cvt_bf16_kernel<<<(int)(WEIGHT_N / 8 / 256), 256, 0, stream>>>(Wq, Wq_h, (int)(WEIGHT_N / 8));
    cvt_bf16_kernel<<<(int)(WEIGHT_N / 8 / 256), 256, 0, stream>>>(Wk, Wk_h, (int)(WEIGHT_N / 8));
    cvt_bf16_kernel<<<(int)(WEIGHT_N / 8 / 256), 256, 0, stream>>>(Wv, Wv_h, (int)(WEIGHT_N / 8));

    // 2) GEMMs (grid: N/256 x M/128)
    dim3 ggrid(Ndim / 256, Mdim / 128);
    float* v_out = out + 2 * (size_t)Bsz * NHEADS * Tlen * HEAD256; // after q,k
    gemm_bf16_wmma<<<ggrid, 256, 0, stream>>>(qs_h, Wq_h, qC, 0);
    gemm_bf16_wmma<<<ggrid, 256, 0, stream>>>(ks_h, Wk_h, kC, 0);
    gemm_bf16_wmma<<<ggrid, 256, 0, stream>>>(vs_h, Wv_h, v_out, 1);

    // 3) fused concat + LN + head-split + RoPE for q and k
    float* q_out = out;
    float* k_out = out + (size_t)Bsz * NHEADS * Tlen * HEAD256;
    ln_rope_kernel<<<Mdim, 256, 0, stream>>>(qC, qg, qb, negator, off, q_out);
    ln_rope_kernel<<<Mdim, 256, 0, stream>>>(kC, kg, kb, negator, off, k_out);
}
